// TemporalMambaSSM_56126632624833
// MI455X (gfx1250) — matrix operand order repaired and checked by reading, not compiled, not run on hardware
//
#include <hip/hip_runtime.h>

// TemporalMambaSSM for gfx1250 (MI455X).
//   1) in_proj GEMM (bf16 WMMA, f32 acc):  xz[M,1024] = x[M,256] @ Wi[1024,256]^T
//   2) depthwise causal conv1d(4)+SiLU in place on xz[:, 0:512]
//   3) x_proj GEMM (WMMA): x_dbl[M,48] = u[M,512] @ Wx[48,512]^T
//   4) fused dt_proj + softplus + selective scan + D-skip + SiLU gate,
//      writes y in place over u (xz[:, 0:512]); res read from xz[:, 512:1024]
//   5) out_proj GEMM (WMMA): out[M,256] = y[M,512] @ Wo[256,512]^T
// M = 784*64 = 50176 rows. Workspace: xz (205.5 MB) + x_dbl (9.6 MB).

#define MROWS (784 * 64)

typedef __attribute__((ext_vector_type(16))) __bf16 v16bf;
typedef __attribute__((ext_vector_type(8)))  float  v8f;

// Native f32->bf16 conversion (lowers to v_cvt_pk_bf16_f32 on gfx1250).
__device__ __forceinline__ void cvt8(v16bf& v, int base, float4 a, float4 b) {
  v[base + 0] = (__bf16)a.x; v[base + 1] = (__bf16)a.y;
  v[base + 2] = (__bf16)a.z; v[base + 3] = (__bf16)a.w;
  v[base + 4] = (__bf16)b.x; v[base + 5] = (__bf16)b.y;
  v[base + 6] = (__bf16)b.z; v[base + 7] = (__bf16)b.w;
}

// C[m, n] = sum_k A[m, k] * W[n, k]   (A row stride lda, C row stride ldc)
// 256 threads = 8 waves arranged WM x WN; each wave owns MT*16 rows x NT*16 cols.
// Per 32-wide k-step: load+convert all MT A frags and NT B frags (one big
// clustered load burst), then issue MT*NT WMMAs back-to-back, so the
// WMMA->VALU WAR hazard (4 co-exec NOPs for bf16) hits once per k-step
// instead of once per fragment.
// grid.x * WM*MT*16 == M, grid.y * WN*NT*16 == N.
template <int WM, int WN, int MT, int NT>
__global__ __launch_bounds__(256) void gemm_bf16_wmma(
    const float* __restrict__ A, const float* __restrict__ W,
    float* __restrict__ C, int K, int lda, int ldc) {
  const int lane = threadIdx.x & 31;
  const int wave = threadIdx.x >> 5;
  const int hi   = lane >> 4;     // 0: lanes 0-15, 1: lanes 16-31
  const int l16  = lane & 15;
  const int wm   = wave % WM;
  const int wn   = wave / WM;
  const int m0   = blockIdx.x * (WM * MT * 16) + wm * (MT * 16);
  const int n0   = blockIdx.y * (WN * NT * 16) + wn * (NT * 16);

  const v8f vz = {0.f, 0.f, 0.f, 0.f, 0.f, 0.f, 0.f, 0.f};
  v8f acc[MT][NT];
#pragma unroll
  for (int i = 0; i < MT; ++i)
#pragma unroll
    for (int j = 0; j < NT; ++j) acc[i][j] = vz;

  for (int k0 = 0; k0 < K; k0 += 32) {
    // A fragments: row = m0 + 16i + l16;
    // K spans {k0+hi*8 .. +7} (elems 0-7) and {k0+16+hi*8 .. +7} (elems 8-15)
    v16bf va[MT];
#pragma unroll
    for (int i = 0; i < MT; ++i) {
      const float* ap = A + (size_t)(m0 + i * 16 + l16) * lda + (k0 + hi * 8);
      cvt8(va[i], 0, *(const float4*)(ap + 0),  *(const float4*)(ap + 4));
      cvt8(va[i], 8, *(const float4*)(ap + 16), *(const float4*)(ap + 20));
    }
    // B fragments: col = n0+16j+l16; K spans {k0+hi*8 .. +7} and {k0+16+hi*8 .. +7}
    v16bf vb[NT];
#pragma unroll
    for (int j = 0; j < NT; ++j) {
      const float* wp = W + (size_t)(n0 + j * 16 + l16) * K + (k0 + hi * 8);
      cvt8(vb[j], 0, *(const float4*)(wp + 0), *(const float4*)(wp + 4));
      cvt8(vb[j], 8, *(const float4*)(wp + 16), *(const float4*)(wp + 20));
    }
    // Matrix core burst: MT*NT WMMAs, no intervening VALU on frag registers.
#pragma unroll
    for (int j = 0; j < NT; ++j)
#pragma unroll
      for (int i = 0; i < MT; ++i)
        acc[i][j] = __builtin_amdgcn_wmma_f32_16x16x32_bf16(
            false, va[i], false, vb[j], (short)0, acc[i][j], false, false);
  }

  // D layout: VGPR r -> row (tile_m + hi*8 + r), lane column n0 + 16j + l16
#pragma unroll
  for (int i = 0; i < MT; ++i) {
    const int mb = m0 + i * 16 + hi * 8;
#pragma unroll
    for (int j = 0; j < NT; ++j) {
      const int nc = n0 + j * 16 + l16;
#pragma unroll
      for (int r = 0; r < 8; ++r)
        C[(size_t)(mb + r) * ldc + nc] = acc[i][j][r];
    }
  }
}

// Depthwise causal conv1d(width 4) over t, then SiLU; in place on xz[:, 0:512].
__global__ __launch_bounds__(512) void conv_silu_inplace(
    float* __restrict__ xz, const float* __restrict__ cw,
    const float* __restrict__ cb) {
  const int b = blockIdx.x;
  const int d = threadIdx.x;
  const float w0 = cw[d * 4 + 0], w1 = cw[d * 4 + 1];
  const float w2 = cw[d * 4 + 2], w3 = cw[d * 4 + 3];
  const float bias = cb[d];
  float x0 = 0.f, x1 = 0.f, x2 = 0.f;  // sliding window (t-3, t-2, t-1)
  float* base = xz + (size_t)b * 64 * 1024 + d;
#pragma unroll 4
  for (int t = 0; t < 64; ++t) {
    const float xin = base[(size_t)t * 1024];
    float v = w0 * x0 + w1 * x1 + w2 * x2 + w3 * xin + bias;
    v = v / (1.f + __expf(-v));
    base[(size_t)t * 1024] = v;
    x0 = x1; x1 = x2; x2 = xin;
  }
}

// Fused dt_proj + softplus + selective scan + D skip + SiLU(res) gate.
// One block per b; thread d owns channel d; x_dbl slab staged in LDS.
__global__ __launch_bounds__(512) void selective_scan(
    float* __restrict__ xz, const float* __restrict__ xdbl,
    const float* __restrict__ dtw, const float* __restrict__ dtb,
    const float* __restrict__ alog, const float* __restrict__ dvec) {
  __shared__ float s[64 * 48];  // 12 KB: per-t [dt(16) | B(16) | C(16)]
  const int b = blockIdx.x;
  const int d = threadIdx.x;
  for (int i = threadIdx.x; i < 64 * 48; i += 512)
    s[i] = xdbl[(size_t)b * 64 * 48 + i];
  __syncthreads();

  float wdt[16], Arow[16], h[16];
#pragma unroll
  for (int r = 0; r < 16; ++r) {
    wdt[r]  = dtw[d * 16 + r];
    Arow[r] = -__expf(alog[d * 16 + r]);
    h[r]    = 0.f;
  }
  const float bdt = dtb[d];
  const float Dd  = dvec[d];
  float* base = xz + (size_t)b * 64 * 1024;

  for (int t = 0; t < 64; ++t) {
    const float* row = &s[t * 48];
    float dt = bdt;
#pragma unroll
    for (int r = 0; r < 16; ++r) dt += wdt[r] * row[r];
    const float delta = (dt > 20.f) ? dt : __logf(1.f + __expf(dt));
    const float ut = base[(size_t)t * 1024 + d];
    const float du = delta * ut;
    float y = 0.f;
#pragma unroll
    for (int n = 0; n < 16; ++n) {
      h[n] = h[n] * __expf(delta * Arow[n]) + du * row[16 + n];
      y += h[n] * row[32 + n];
    }
    y += ut * Dd;
    const float rv = base[(size_t)t * 1024 + 512 + d];
    y *= rv / (1.f + __expf(-rv));
    base[(size_t)t * 1024 + d] = y;  // overwrite u with gated y
  }
}

extern "C" void kernel_launch(void* const* d_in, const int* in_sizes, int n_in,
                              void* d_out, int out_size, void* d_ws, size_t ws_size,
                              hipStream_t stream) {
  (void)in_sizes; (void)n_in; (void)out_size; (void)ws_size;
  const float* x    = (const float*)d_in[0];
  const float* wi   = (const float*)d_in[1];
  const float* cw   = (const float*)d_in[2];
  const float* cb   = (const float*)d_in[3];
  const float* wx   = (const float*)d_in[4];
  const float* wdt  = (const float*)d_in[5];
  const float* bdt  = (const float*)d_in[6];
  const float* alog = (const float*)d_in[7];
  const float* dvec = (const float*)d_in[8];
  const float* wo   = (const float*)d_in[9];
  float* out = (float*)d_out;

  float* xz = (float*)d_ws;                    // [M, 1024]
  float* xd = xz + (size_t)MROWS * 1024;       // [M, 48]

  // 1) in_proj: xz = x @ Wi^T   (M=50176, K=256, N=1024); 128x128 block tile
  gemm_bf16_wmma<4, 2, 2, 4><<<dim3(392, 8), 256, 0, stream>>>(
      x, wi, xz, 256, 256, 1024);
  // 2) conv + SiLU in place on u half
  conv_silu_inplace<<<784, 512, 0, stream>>>(xz, cw, cb);
  // 3) x_proj: x_dbl = u @ Wx^T  (K=512, N=48); 256x48 block tile
  gemm_bf16_wmma<8, 1, 2, 3><<<dim3(196, 1), 256, 0, stream>>>(
      xz, wx, xd, 512, 1024, 48);
  // 4) scan (fused dt_proj/softplus/SSM/skip/gate), y overwrites u in xz
  selective_scan<<<784, 512, 0, stream>>>(xz, xd, wdt, bdt, alog, dvec);
  // 5) out_proj: out = y @ Wo^T (K=512, N=256); 128x128 block tile
  gemm_bf16_wmma<4, 2, 2, 4><<<dim3(392, 2), 256, 0, stream>>>(
      xz, wo, out, 512, 1024, 256);
}
